// Net_46995532153129
// MI455X (gfx1250) — matrix operand order repaired and checked by reading, not compiled, not run on hardware
//
#include <hip/hip_runtime.h>
#include <math.h>

// ---------------- problem constants (from the reference) ----------------
constexpr int B_   = 20;
constexpr int N_   = 1000;
constexpr int K_   = 15;
constexpr int L_   = 7;
constexpr int KS_  = 5;
constexpr int FN_  = 10;
constexpr int N2_  = 200;   // N/5
constexpr int NCLS_= 40;

typedef __attribute__((ext_vector_type(16))) _Float16 v16h;
typedef __attribute__((ext_vector_type(8)))  float    v8f;

// ---------------- small helpers ----------------
__device__ __forceinline__ float sigmoidf_(float x) { return 1.f / (1.f + expf(-x)); }

__device__ __forceinline__ v8f wmma16(v16h a, v16h b, v8f c) {
  // D = A(16x32 f16) * B(32x16 f16) + C(16x16 f32)
  return __builtin_amdgcn_wmma_f32_16x16x32_f16(false, a, false, b, (short)0, c, false, false);
}

// wave-local LDS visibility fence (staging buffers are per-wave, no block barrier needed)
__device__ __forceinline__ void wave_lds_fence() {
  asm volatile("s_wait_dscnt 0" ::: "memory");
}

// store two f32 values as a packed pair of f16 (one 32-bit LDS store)
__device__ __forceinline__ void store_h2(_Float16* p, float a, float b) {
  union { unsigned u; _Float16 h[2]; } t;
  t.h[0] = (_Float16)a; t.h[1] = (_Float16)b;
  *reinterpret_cast<unsigned*>(p) = t.u;
}

// A fragment: 16x32 f16 row-major tile in LDS (ISA 7.12.2 16-bit A layout).
// Per lane the 16 halves form two contiguous 16B runs -> 2x ds_load_b128.
__device__ __forceinline__ v16h frag_a(const _Float16* s, int ld) {
  const int lane = threadIdx.x & 31;
  const int mm = lane & 15;
  const int kh = (lane >> 4) << 3;          // 0 or 8
  v16h a;
#pragma unroll
  for (int v = 0; v < 8; ++v) {
    const int kb = ((v & 4) << 2) + kh + ((v & 3) << 1);  // (v<4?0:16)+kh+2*(v%4)
    a[2 * v]     = s[mm * ld + kb];
    a[2 * v + 1] = s[mm * ld + kb + 1];
  }
  return a;
}

// B fragment from PRE-SWIZZLED weight storage: fragment = 512 contiguous halves,
// lane's 16 halves contiguous and 32B aligned -> one v16h LDS load (2x b128).
__device__ __forceinline__ v16h frag_b_swz(const _Float16* fragbase) {
  const int lane = threadIdx.x & 31;
  return *reinterpret_cast<const v16h*>(fragbase + lane * 16);
}

// Stage a KrowsxNcols f32 weight matrix into fragment-native swizzled f16 layout,
// zero-padded to Kpad x Npad. Fragment order: fk-major, fn-minor; within a
// fragment: dst[lane*16 + e] = W[fk*32 + (e>=8?16:0) + (lane>=16?8:0) + (e&7)][fn*16 + (lane&15)].
__device__ __forceinline__ void stage_b_swz(_Float16* dst, const float* W,
                                            int Krows, int Ncols, int Kpad, int Npad) {
  const int nfn = Npad >> 4;
  const int total = Kpad * Npad;
  for (int t = threadIdx.x; t < total; t += 256) {
    const int frag = t >> 9;          // 512 halves per fragment
    const int r    = t & 511;
    const int lane = r >> 4, e = r & 15;
    const int fk = frag / nfn, fn = frag - fk * nfn;
    const int k = (fk << 5) + ((e >> 3) << 4) + ((lane >> 4) << 3) + (e & 7);
    const int c = (fn << 4) + (lane & 15);
    dst[t] = (_Float16)((k < Krows && c < Ncols) ? W[k * Ncols + c] : 0.f);
  }
}

// C fragment seeded with a per-column bias (C element (M=r+8*hi, N=lane&15))
__device__ __forceinline__ v8f frag_c_bias(const float* bias, int colbase, int dim) {
  const int lane = threadIdx.x & 31;
  const int col = colbase + (lane & 15);
  const float v = (col < dim) ? bias[col] : 0.f;
  v8f c;
#pragma unroll
  for (int r = 0; r < 8; ++r) c[r] = v;
  return c;
}

// store D fragment with ReLU into a row-major f16 LDS tile at column colbase
__device__ __forceinline__ void store_relu_f16(v8f d, _Float16* tile, int ld, int colbase) {
  const int lane = threadIdx.x & 31;
  const int nn_ = lane & 15;
  const int mo  = (lane >> 4) << 3;
#pragma unroll
  for (int r = 0; r < 8; ++r)
    tile[(mo + r) * ld + colbase + nn_] = (_Float16)fmaxf(d[r], 0.f);
}

// per-column max of relu(D) over rows 0..14 (row 15 is the zero-pad edge)
__device__ __forceinline__ float colmax15_relu(v8f d) {
  const int lane = threadIdx.x & 31;
  const int mo = (lane >> 4) << 3;
  float m = 0.f;                       // relu >= 0, so 0 is a valid identity here
#pragma unroll
  for (int r = 0; r < 8; ++r)
    if (mo + r < 15) m = fmaxf(m, fmaxf(d[r], 0.f));
  m = fmaxf(m, __shfl_xor(m, 16, 32));
  return m;
}

// ---------------- closed-form symmetric 3x3 eigendecomposition ----------------
__device__ __forceinline__ void eigvec3(float a00, float a01, float a02,
                                        float a11, float a12, float a22,
                                        float lam, float v[3]) {
  const float r0x = a00 - lam, r0y = a01,       r0z = a02;
  const float r1x = a01,       r1y = a11 - lam, r1z = a12;
  const float r2x = a02,       r2y = a12,       r2z = a22 - lam;
  const float cx0 = r0y * r1z - r0z * r1y, cy0 = r0z * r1x - r0x * r1z, cz0 = r0x * r1y - r0y * r1x;
  const float cx1 = r0y * r2z - r0z * r2y, cy1 = r0z * r2x - r0x * r2z, cz1 = r0x * r2y - r0y * r2x;
  const float cx2 = r1y * r2z - r1z * r2y, cy2 = r1z * r2x - r1x * r2z, cz2 = r1x * r2y - r1y * r2x;
  const float n0 = cx0 * cx0 + cy0 * cy0 + cz0 * cz0;
  const float n1 = cx1 * cx1 + cy1 * cy1 + cz1 * cz1;
  const float n2 = cx2 * cx2 + cy2 * cy2 + cz2 * cz2;
  float cx = cx0, cy = cy0, cz = cz0, nb = n0;
  if (n1 > nb) { cx = cx1; cy = cy1; cz = cz1; nb = n1; }
  if (n2 > nb) { cx = cx2; cy = cy2; cz = cz2; nb = n2; }
  if (nb < 1e-30f) { v[0] = 1.f; v[1] = 0.f; v[2] = 0.f; return; }
  const float inr = rsqrtf(nb);
  v[0] = cx * inr; v[1] = cy * inr; v[2] = cz * inr;
}

__device__ __forceinline__ void eig3(float a00, float a01, float a02,
                                     float a11, float a12, float a22, float R[9]) {
  const float q = (a00 + a11 + a22) * (1.f / 3.f);
  const float b00 = a00 - q, b11 = a11 - q, b22 = a22 - q;
  const float p2 = b00 * b00 + b11 * b11 + b22 * b22 +
                   2.f * (a01 * a01 + a02 * a02 + a12 * a12);
  if (p2 < 1e-24f) {  // (near) diagonal / isotropic: identity frame
    R[0] = 1.f; R[1] = 0.f; R[2] = 0.f;
    R[3] = 0.f; R[4] = 1.f; R[5] = 0.f;
    R[6] = 0.f; R[7] = 0.f; R[8] = 1.f;
    return;
  }
  const float p  = sqrtf(p2 * (1.f / 6.f));
  const float ip = 1.f / p;
  const float d00 = b00 * ip, d01 = a01 * ip, d02 = a02 * ip;
  const float d11 = b11 * ip, d12 = a12 * ip, d22 = b22 * ip;
  const float det = d00 * (d11 * d22 - d12 * d12)
                  - d01 * (d01 * d22 - d12 * d02)
                  + d02 * (d01 * d12 - d11 * d02);
  const float rr  = fminf(fmaxf(det * 0.5f, -1.f), 1.f);
  const float phi = acosf(rr) * (1.f / 3.f);
  const float lam2 = q + 2.f * p * cosf(phi);                       // largest
  const float lam0 = q + 2.f * p * cosf(phi + 2.0943951023931953f); // smallest
  float v0[3], v2[3];
  eigvec3(a00, a01, a02, a11, a12, a22, lam0, v0);
  eigvec3(a00, a01, a02, a11, a12, a22, lam2, v2);
  float v1x = v2[1] * v0[2] - v2[2] * v0[1];
  float v1y = v2[2] * v0[0] - v2[0] * v0[2];
  float v1z = v2[0] * v0[1] - v2[1] * v0[0];
  const float nsq = v1x * v1x + v1y * v1y + v1z * v1z;
  const float inr = (nsq > 1e-30f) ? rsqrtf(nsq) : 0.f;
  v1x *= inr; v1y *= inr; v1z *= inr;
  // columns = eigenvectors, ascending eigenvalue (like jnp.linalg.eigh); R[d*3+e]
  R[0] = v0[0]; R[1] = v1x; R[2] = v2[0];
  R[3] = v0[1]; R[4] = v1y; R[5] = v2[1];
  R[6] = v0[2]; R[7] = v1z; R[8] = v2[2];
}

// =======================================================================
// Kernel 1: per-cloud knn (top-15), local PCA frame, spline conv -> f3d
// grid = B_, block = 256; pos + Wsp staged in LDS
// =======================================================================
__global__ void __launch_bounds__(256) k_knn_frame_spline(
    const float* __restrict__ pos, const float* __restrict__ Wsp,
    int* __restrict__ idxg, float* __restrict__ Rg, float* __restrict__ f3dg)
{
  __shared__ float spos[N_ * 3];
  __shared__ float sW[KS_ * KS_ * KS_ * FN_];
  const int b = blockIdx.x, tid = threadIdx.x;
  for (int t = tid; t < N_ * 3; t += 256) spos[t] = pos[(size_t)b * N_ * 3 + t];
  for (int t = tid; t < KS_ * KS_ * KS_ * FN_; t += 256) sW[t] = Wsp[t];
  __syncthreads();

  for (int i = tid; i < N_; i += 256) {
    const float px = spos[3 * i], py = spos[3 * i + 1], pz = spos[3 * i + 2];

    // --- register-resident sorted top-K (ascending distance) ---
    float bd[K_]; int bi_[K_];
#pragma unroll
    for (int k = 0; k < K_; ++k) { bd[k] = 3.0e38f; bi_[k] = 0; }
    for (int j = 0; j < N_; ++j) {
      if (j == i) continue;  // loop=False / +1e9 on diagonal
      const float dx = spos[3 * j] - px, dy = spos[3 * j + 1] - py, dz = spos[3 * j + 2] - pz;
      const float d = dx * dx + dy * dy + dz * dz;
      if (d < bd[K_ - 1]) {
        bd[K_ - 1] = d; bi_[K_ - 1] = j;
#pragma unroll
        for (int s = K_ - 1; s > 0; --s) {
          if (bd[s] < bd[s - 1]) {
            const float td = bd[s]; bd[s] = bd[s - 1]; bd[s - 1] = td;
            const int   ti = bi_[s]; bi_[s] = bi_[s - 1]; bi_[s - 1] = ti;
          }
        }
      }
    }
    const int base = (b * N_ + i) * K_;
#pragma unroll
    for (int k = 0; k < K_; ++k) idxg[base + k] = bi_[k];

    // --- covariance over the L nearest, analytic eigh -> frame R ---
    float c00 = 0, c01 = 0, c02 = 0, c11 = 0, c12 = 0, c22 = 0;
#pragma unroll
    for (int k = 0; k < L_; ++k) {
      const int j = bi_[k];
      const float rx = spos[3 * j] - px, ry = spos[3 * j + 1] - py, rz = spos[3 * j + 2] - pz;
      c00 += rx * rx; c01 += rx * ry; c02 += rx * rz;
      c11 += ry * ry; c12 += ry * rz; c22 += rz * rz;
    }
    float Rv[9];
    eig3(c00, c01, c02, c11, c12, c22, Rv);

    // --- spline conv: pass 1, max |rel_rot| (read idx back from L2 to keep loops rolled) ---
    float mx = 0.f;
    for (int k = 0; k < K_; ++k) {
      const int j = idxg[base + k];
      const float rx = spos[3 * j] - px, ry = spos[3 * j + 1] - py, rz = spos[3 * j + 2] - pz;
      const float rr0 = rx * Rv[0] + ry * Rv[3] + rz * Rv[6];
      const float rr1 = rx * Rv[1] + ry * Rv[4] + rz * Rv[7];
      const float rr2 = rx * Rv[2] + ry * Rv[5] + rz * Rv[8];
      mx = fmaxf(mx, fmaxf(fabsf(rr0), fmaxf(fabsf(rr1), fabsf(rr2))));
    }
    mx += 1e-8f;
    const float inv2m = 0.5f / mx;

    // --- pass 2: trilinear B-spline accumulate ---
    float acc[FN_];
#pragma unroll
    for (int c = 0; c < FN_; ++c) acc[c] = 0.f;
    for (int k = 0; k < K_; ++k) {
      const int j = idxg[base + k];
      const float rx = spos[3 * j] - px, ry = spos[3 * j + 1] - py, rz = spos[3 * j + 2] - pz;
      const float rr0 = rx * Rv[0] + ry * Rv[3] + rz * Rv[6];
      const float rr1 = rx * Rv[1] + ry * Rv[4] + rz * Rv[7];
      const float rr2 = rx * Rv[2] + ry * Rv[5] + rz * Rv[8];
      const float u0 = fminf(fmaxf(rr0 * inv2m + 0.5f, 0.f), 0.999999f);
      const float u1 = fminf(fmaxf(rr1 * inv2m + 0.5f, 0.f), 0.999999f);
      const float u2 = fminf(fmaxf(rr2 * inv2m + 0.5f, 0.f), 0.999999f);
      const float t0 = u0 * (KS_ - 1), t1 = u1 * (KS_ - 1), t2 = u2 * (KS_ - 1);
      const int ix = min((int)t0, KS_ - 2), iy = min((int)t1, KS_ - 2), iz = min((int)t2, KS_ - 2);
      const float f0 = t0 - (float)ix, f1 = t1 - (float)iy, f2 = t2 - (float)iz;
#pragma unroll
      for (int dx = 0; dx < 2; ++dx)
#pragma unroll
        for (int dy = 0; dy < 2; ++dy)
#pragma unroll
          for (int dz = 0; dz < 2; ++dz) {
            const float w = (dx ? f0 : 1.f - f0) * (dy ? f1 : 1.f - f1) * (dz ? f2 : 1.f - f2);
            const int flat = (((ix + dx) * KS_) + (iy + dy)) * KS_ + (iz + dz);
            const float* wr = &sW[flat * FN_];
#pragma unroll
            for (int c = 0; c < FN_; ++c) acc[c] += w * wr[c];
          }
    }
    const float invK = 1.f / (float)K_;
#pragma unroll
    for (int c = 0; c < FN_; ++c) f3dg[(b * N_ + i) * FN_ + c] = sigmoidf_(acc[c] * invK);
#pragma unroll
    for (int e = 0; e < 9; ++e) Rg[(b * N_ + i) * 9 + e] = Rv[e];
  }
}

// =======================================================================
// Kernel 2: farthest point sampling (exact lax.scan semantics) + knn over
// the 200 sampled points. grid = B_, block = 256.
// =======================================================================
__global__ void __launch_bounds__(256) k_fps_knn2(
    const float* __restrict__ pos, int* __restrict__ sidxg, int* __restrict__ idx2g)
{
  __shared__ float spos[N_ * 3];
  __shared__ float dmin[N_];
  __shared__ float p2s[N2_ * 3];
  __shared__ float redv[256];
  __shared__ int   redi[256];
  const int b = blockIdx.x, tid = threadIdx.x;
  for (int t = tid; t < N_ * 3; t += 256) spos[t] = pos[(size_t)b * N_ * 3 + t];
  for (int t = tid; t < N_; t += 256) dmin[t] = 3.0e38f;
  __syncthreads();

  int last = 0;
  for (int step = 0; step < N2_; ++step) {
    if (tid == 0) sidxg[b * N2_ + step] = last;     // sel[t] = last (pre-update)
    if (tid < 3)  p2s[step * 3 + tid] = spos[last * 3 + tid];
    const float lx = spos[3 * last], ly = spos[3 * last + 1], lz = spos[3 * last + 2];
    for (int j = tid; j < N_; j += 256) {
      const float dx = spos[3 * j] - lx, dy = spos[3 * j + 1] - ly, dz = spos[3 * j + 2] - lz;
      dmin[j] = fminf(dmin[j], dx * dx + dy * dy + dz * dz);
    }
    __syncthreads();
    // first-max argmax (ties -> lowest index)
    float bv = -1.f; int bidx = N_;
    for (int j = tid; j < N_; j += 256) {
      const float v = dmin[j];
      if (v > bv) { bv = v; bidx = j; }
    }
    redv[tid] = bv; redi[tid] = bidx;
    __syncthreads();
    for (int s = 128; s > 0; s >>= 1) {
      if (tid < s) {
        if (redv[tid + s] > redv[tid] ||
            (redv[tid + s] == redv[tid] && redi[tid + s] < redi[tid])) {
          redv[tid] = redv[tid + s]; redi[tid] = redi[tid + s];
        }
      }
      __syncthreads();
    }
    last = redi[0];
    __syncthreads();
  }

  // knn over sampled cloud (200 points -> one per thread)
  if (tid < N2_) {
    const int i = tid;
    const float px = p2s[3 * i], py = p2s[3 * i + 1], pz = p2s[3 * i + 2];
    float bd[K_]; int bi_[K_];
#pragma unroll
    for (int k = 0; k < K_; ++k) { bd[k] = 3.0e38f; bi_[k] = 0; }
    for (int j = 0; j < N2_; ++j) {
      if (j == i) continue;
      const float dx = p2s[3 * j] - px, dy = p2s[3 * j + 1] - py, dz = p2s[3 * j + 2] - pz;
      const float d = dx * dx + dy * dy + dz * dz;
      if (d < bd[K_ - 1]) {
        bd[K_ - 1] = d; bi_[K_ - 1] = j;
#pragma unroll
        for (int s = K_ - 1; s > 0; --s) {
          if (bd[s] < bd[s - 1]) {
            const float td = bd[s]; bd[s] = bd[s - 1]; bd[s - 1] = td;
            const int   ti = bi_[s]; bi_[s] = bi_[s - 1]; bi_[s - 1] = ti;
          }
        }
      }
    }
#pragma unroll
    for (int k = 0; k < K_; ++k) idx2g[(b * N2_ + i) * K_ + k] = bi_[k];
  }
}

// =======================================================================
// Kernel 3: dense block 1 via WMMA. One wave per point: A = 16x64 edge tile
// (15 edges + zero pad), W1 padded 64x32, W2 padded 32x32, f32 accumulate.
// Weights pre-swizzled to fragment layout (vector LDS loads).
// grid = B_*N_/8 = 2500, block = 256 (8 waves).
// =======================================================================
__global__ void __launch_bounds__(256) k_dense1(
    const float* __restrict__ pos, const float* __restrict__ f3dg,
    const float* __restrict__ Rg, const int* __restrict__ idxg,
    const float* __restrict__ W1, const float* __restrict__ b1,
    const float* __restrict__ W2, const float* __restrict__ b2,
    float* __restrict__ fddg)
{
  __shared__ __align__(32) _Float16 W1f[4 * 512];  // frags (fk 0..1) x (fn 0..1)
  __shared__ __align__(32) _Float16 W2f[2 * 512];  // frags (fk 0)    x (fn 0..1)
  __shared__ float b1s[32], b2s[32];
  __shared__ __align__(32) _Float16 eS[8 * 16 * 64];
  __shared__ __align__(32) _Float16 hS[8 * 16 * 32];
  const int tid = threadIdx.x;
  stage_b_swz(W1f, W1, 33, 20, 64, 32);
  stage_b_swz(W2f, W2, 20, 20, 32, 32);
  if (tid < 32) {
    b1s[tid] = (tid < 20) ? b1[tid] : 0.f;
    b2s[tid] = (tid < 20) ? b2[tid] : 0.f;
  }
  __syncthreads();

  const int lane = tid & 31, wv = tid >> 5;
  const int pt = blockIdx.x * 8 + wv;
  const int b = pt / N_, i = pt - b * N_;
  _Float16* e = &eS[wv * 16 * 64];
  _Float16* h = &hS[wv * 16 * 32];
  const int row = lane & 15, cg = lane >> 4;   // two lanes per row; cg = column half
  const int cbase = cg * 32;

  if (row < 15) {
    const int j = idxg[(b * N_ + i) * K_ + row];
    const float* fiP = &f3dg[(b * N_ + i) * FN_];
    const float* fjP = &f3dg[(b * N_ + j) * FN_];
    const float rx = pos[((size_t)b * N_ + j) * 3 + 0] - pos[((size_t)b * N_ + i) * 3 + 0];
    const float ry = pos[((size_t)b * N_ + j) * 3 + 1] - pos[((size_t)b * N_ + i) * 3 + 1];
    const float rz = pos[((size_t)b * N_ + j) * 3 + 2] - pos[((size_t)b * N_ + i) * 3 + 2];
    const float* Rr = &Rg[(b * N_ + i) * 9];
    const float rr0 = rx * Rr[0] + ry * Rr[3] + rz * Rr[6];
    const float rr1 = rx * Rr[1] + ry * Rr[4] + rz * Rr[7];
    const float rr2 = rx * Rr[2] + ry * Rr[5] + rz * Rr[8];
    auto edge_val = [&](int c) -> float {
      if (c < 10)  return fiP[c];
      if (c < 20)  return fjP[c - 10];
      if (c < 30)  return fjP[c - 20] - fiP[c - 20];
      if (c == 30) return rr0;
      if (c == 31) return rr1;
      if (c == 32) return rr2;
      return 0.f;
    };
    for (int c0 = cbase; c0 < cbase + 32; c0 += 2)
      store_h2(&e[row * 64 + c0], edge_val(c0), edge_val(c0 + 1));
  } else {
    for (int c0 = cbase; c0 < cbase + 32; c0 += 2)
      store_h2(&e[row * 64 + c0], 0.f, 0.f);
  }
  wave_lds_fence();

  const v16h a0 = frag_a(e, 64);        // K = 0..31
  const v16h a1 = frag_a(e + 32, 64);   // K = 32..63 (only K=32 nonzero)
#pragma unroll
  for (int t = 0; t < 2; ++t) {
    const v16h blo = frag_b_swz(&W1f[(0 * 2 + t) * 512]);  // K 0..31,  cols 16t..
    const v16h bhi = frag_b_swz(&W1f[(1 * 2 + t) * 512]);  // K 32..63, cols 16t..
    v8f c = frag_c_bias(b1s, t * 16, 20);
    c = wmma16(a0, blo, c);
    c = wmma16(a1, bhi, c);
    store_relu_f16(c, h, 32, t * 16);   // relu(e@W1+b1) -> 16x32 f16
  }
  wave_lds_fence();

  const v16h ah = frag_a(h, 32);        // K = 0..31 (cols 20..31 are zeros)
#pragma unroll
  for (int t = 0; t < 2; ++t) {
    const v16h bb = frag_b_swz(&W2f[t * 512]);
    v8f c = frag_c_bias(b2s, t * 16, 20);
    c = wmma16(ah, bb, c);
    const float mxv = colmax15_relu(c); // relu then max over the 15 real edges
    const int col = t * 16 + (lane & 15);
    if (lane < 16 && col < 20)
      fddg[(b * N_ + i) * 20 + col] = sigmoidf_(mxv);
  }
}

// =======================================================================
// Kernel 4: dense block 2 via WMMA (60->64 in, 64 hidden, 32 out).
// grid = B_*N2_/8 = 500, block = 256.
// =======================================================================
__global__ void __launch_bounds__(256) k_dense2(
    const float* __restrict__ fddg, const int* __restrict__ sidxg,
    const int* __restrict__ idx2g,
    const float* __restrict__ W3, const float* __restrict__ b3,
    const float* __restrict__ W4, const float* __restrict__ b4,
    float* __restrict__ f22g)
{
  __shared__ __align__(32) _Float16 W3f[8 * 512];  // frags (fk 0..1) x (fn 0..3)
  __shared__ __align__(32) _Float16 W4f[4 * 512];  // frags (fk 0..1) x (fn 0..1)
  __shared__ float b3s[64], b4s[32];
  __shared__ __align__(32) _Float16 eS[8 * 16 * 64];
  __shared__ __align__(32) _Float16 hS[8 * 16 * 64];
  const int tid = threadIdx.x;
  stage_b_swz(W3f, W3, 60, 64, 64, 64);
  stage_b_swz(W4f, W4, 64, 32, 64, 32);
  if (tid < 64) b3s[tid] = b3[tid];
  if (tid < 32) b4s[tid] = b4[tid];
  __syncthreads();

  const int lane = tid & 31, wv = tid >> 5;
  const int pt = blockIdx.x * 8 + wv;
  const int b = pt / N2_, mpt = pt - b * N2_;
  _Float16* e = &eS[wv * 16 * 64];
  _Float16* h = &hS[wv * 16 * 64];
  const int row = lane & 15, cg = lane >> 4;
  const int cbase = cg * 32;

  const int gi = sidxg[b * N2_ + mpt];
  const float* fiP = &fddg[(b * N_ + gi) * 20];
  if (row < 15) {
    const int j2 = idx2g[(b * N2_ + mpt) * K_ + row];
    const int gj = sidxg[b * N2_ + j2];
    const float* fjP = &fddg[(b * N_ + gj) * 20];
    auto edge_val = [&](int c) -> float {
      if (c < 20) return fiP[c];
      if (c < 40) return fjP[c - 20];
      if (c < 60) return fjP[c - 40] - fiP[c - 40];
      return 0.f;
    };
    for (int c0 = cbase; c0 < cbase + 32; c0 += 2)
      store_h2(&e[row * 64 + c0], edge_val(c0), edge_val(c0 + 1));
  } else {
    for (int c0 = cbase; c0 < cbase + 32; c0 += 2)
      store_h2(&e[row * 64 + c0], 0.f, 0.f);
  }
  wave_lds_fence();

  const v16h a0 = frag_a(e, 64);
  const v16h a1 = frag_a(e + 32, 64);
#pragma unroll
  for (int t = 0; t < 4; ++t) {
    const v16h blo = frag_b_swz(&W3f[(0 * 4 + t) * 512]);
    const v16h bhi = frag_b_swz(&W3f[(1 * 4 + t) * 512]);
    v8f c = frag_c_bias(b3s, t * 16, 64);
    c = wmma16(a0, blo, c);
    c = wmma16(a1, bhi, c);
    store_relu_f16(c, h, 64, t * 16);   // relu(e@W3+b3) -> 16x64 f16
  }
  wave_lds_fence();

  const v16h h0 = frag_a(h, 64);
  const v16h h1 = frag_a(h + 32, 64);
#pragma unroll
  for (int t = 0; t < 2; ++t) {
    const v16h blo = frag_b_swz(&W4f[(0 * 2 + t) * 512]);
    const v16h bhi = frag_b_swz(&W4f[(1 * 2 + t) * 512]);
    v8f c = frag_c_bias(b4s, t * 16, 32);
    c = wmma16(h0, blo, c);
    c = wmma16(h1, bhi, c);
    const float mxv = colmax15_relu(c);
    const int col = t * 16 + (lane & 15);
    if (lane < 16)
      f22g[(b * N2_ + mpt) * 32 + col] = sigmoidf_(mxv);
  }
}

// =======================================================================
// Kernel 5: head — mean over sampled points, elu dense, log-softmax.
// Tiny (B=20), single block in f32.
// =======================================================================
__global__ void __launch_bounds__(256) k_head(
    const float* __restrict__ f22g,
    const float* __restrict__ Wn1, const float* __restrict__ bn1,
    const float* __restrict__ Wn2, const float* __restrict__ bn2,
    float* __restrict__ out)
{
  __shared__ float ysS[B_ * 32];
  __shared__ float y1S[B_ * 256];
  __shared__ float lgS[B_ * NCLS_];
  const int tid = threadIdx.x;
  for (int o = tid; o < B_ * 32; o += 256) {
    const int b = o >> 5, c = o & 31;
    float s = 0.f;
    for (int t = 0; t < N2_; ++t) s += f22g[(b * N2_ + t) * 32 + c];
    ysS[o] = s * (1.f / (float)N2_);
  }
  __syncthreads();
  for (int o = tid; o < B_ * 256; o += 256) {
    const int b = o >> 8, c = o & 255;
    float s = bn1[c];
    for (int k = 0; k < 32; ++k) s += ysS[b * 32 + k] * Wn1[k * 256 + c];
    y1S[o] = (s > 0.f) ? s : (expf(s) - 1.f);   // elu, alpha=1
  }
  __syncthreads();
  for (int o = tid; o < B_ * NCLS_; o += 256) {
    const int b = o / NCLS_, c = o - b * NCLS_;
    float s = bn2[c];
    for (int k = 0; k < 256; ++k) s += y1S[b * 256 + k] * Wn2[k * NCLS_ + c];
    lgS[o] = s;
  }
  __syncthreads();
  if (tid < B_) {
    const int b = tid;
    float mx = -3.0e38f;
    for (int c = 0; c < NCLS_; ++c) mx = fmaxf(mx, lgS[b * NCLS_ + c]);
    float s = 0.f;
    for (int c = 0; c < NCLS_; ++c) s += expf(lgS[b * NCLS_ + c] - mx);
    const float lse = mx + logf(s);
    for (int c = 0; c < NCLS_; ++c) out[b * NCLS_ + c] = lgS[b * NCLS_ + c] - lse;
  }
}

// =======================================================================
extern "C" void kernel_launch(void* const* d_in, const int* in_sizes, int n_in,
                              void* d_out, int out_size, void* d_ws, size_t ws_size,
                              hipStream_t stream) {
  (void)in_sizes; (void)n_in; (void)out_size; (void)ws_size;
  const float* pos = (const float*)d_in[0];
  const float* Wsp = (const float*)d_in[1];
  const float* W1  = (const float*)d_in[2];  const float* b1  = (const float*)d_in[3];
  const float* W2  = (const float*)d_in[4];  const float* b2  = (const float*)d_in[5];
  const float* W3  = (const float*)d_in[6];  const float* b3  = (const float*)d_in[7];
  const float* W4  = (const float*)d_in[8];  const float* b4  = (const float*)d_in[9];
  const float* Wn1 = (const float*)d_in[10]; const float* bn1 = (const float*)d_in[11];
  const float* Wn2 = (const float*)d_in[12]; const float* bn2 = (const float*)d_in[13];
  float* out = (float*)d_out;

  char* w = (char*)d_ws;
  size_t off = 0;
  auto take = [&](size_t bytes) {
    const size_t o = off;
    off += (bytes + 255) & ~(size_t)255;
    return o;
  };
  int*   idxg  = (int*)  (w + take((size_t)B_ * N_  * K_ * 4));
  float* Rg    = (float*)(w + take((size_t)B_ * N_  * 9  * 4));
  float* f3dg  = (float*)(w + take((size_t)B_ * N_  * FN_ * 4));
  float* fddg  = (float*)(w + take((size_t)B_ * N_  * 20 * 4));
  int*   sidxg = (int*)  (w + take((size_t)B_ * N2_ * 4));
  int*   idx2g = (int*)  (w + take((size_t)B_ * N2_ * K_ * 4));
  float* f22g  = (float*)(w + take((size_t)B_ * N2_ * 32 * 4));

  k_knn_frame_spline<<<B_, 256, 0, stream>>>(pos, Wsp, idxg, Rg, f3dg);
  k_fps_knn2<<<B_, 256, 0, stream>>>(pos, sidxg, idx2g);
  k_dense1<<<(B_ * N_) / 8, 256, 0, stream>>>(pos, f3dg, Rg, idxg, W1, b1, W2, b2, fddg);
  k_dense2<<<(B_ * N2_) / 8, 256, 0, stream>>>(fddg, sidxg, idx2g, W3, b3, W4, b4, f22g);
  k_head<<<1, 256, 0, stream>>>(f22g, Wn1, bn1, Wn2, bn2, out);
}
